// ODEgnn_33071248179559
// MI455X (gfx1250) — matrix operand order repaired and checked by reading, not compiled, not run on hardware
//
#include <hip/hip_runtime.h>
#include <stdint.h>

// ---------------------------------------------------------------------------
// GNN ODE pipeline for MI455X (gfx1250, wave32, WMMA 16x16x32 f16).
// Memory-bound workload (~3.5GB traffic vs ~97 GFLOP) -> f16 activations,
// fused A-fragment generation, async LDS-staged packed weights, f32-atomic
// scatter. Softplus uses HW v_exp/v_log transcendentals (co-execute with WMMA).
// ---------------------------------------------------------------------------

#define HID 128

typedef __attribute__((ext_vector_type(16))) _Float16 v16h;
typedef __attribute__((ext_vector_type(8)))  _Float16 v8h;
typedef __attribute__((ext_vector_type(8)))  float    v8f;
typedef __attribute__((ext_vector_type(4)))  int      i4;

union AFrag { v16h v; uint4 q[2]; unsigned int u[8]; _Float16 h[16]; };
union H8    { uint4 q; v8h h; };
union H4    { uint2 q; _Float16 h[4]; };

// Fast, stable softplus: max(z,0) + log(1 + exp(-|z|)). v_exp_f32 + v_log_f32
// are TRANS32 ops on CDNA5 and co-execute with the WMMA pipe.
__device__ __forceinline__ float softplusf(float z) {
    return fmaxf(z, 0.0f) + __logf(1.0f + __expf(-fabsf(z)));
}

__device__ __forceinline__ unsigned int pack2f16(float a, float b) {
    union { _Float16 h[2]; unsigned int u; } p;
    p.h[0] = (_Float16)a; p.h[1] = (_Float16)b;
    return p.u;
}

// ---------------------------------------------------------------------------
// Async (ASYNCcnt-tracked) global->LDS staging with synchronous fallback.
// Builtin signature (from hipcc diagnostic): (int4 AS1*, int4 AS3*, Ii, Ii).
// ---------------------------------------------------------------------------
#if defined(__gfx1250__) && __has_builtin(__builtin_amdgcn_global_load_async_to_lds_b128)
#define HAVE_ASYNC_LDS 1
#endif

__device__ __forceinline__ void stage_lds(unsigned int* sW, const unsigned int* Wp,
                                          int nwords) {
#ifdef HAVE_ASYNC_LDS
    unsigned int* wp = const_cast<unsigned int*>(Wp);
    for (int i = threadIdx.x * 4; i < nwords; i += 256 * 4) {
        __builtin_amdgcn_global_load_async_to_lds_b128(
            (__attribute__((address_space(1))) i4*)(wp + i),
            (__attribute__((address_space(3))) i4*)(sW + i),
            0, 0);
    }
#if __has_builtin(__builtin_amdgcn_s_wait_asynccnt)
    __builtin_amdgcn_s_wait_asynccnt(0);
#else
    asm volatile("s_wait_asynccnt 0" ::: "memory");
#endif
#else
    for (int i = threadIdx.x; i < nwords; i += 256) sW[i] = Wp[i];
#endif
}

// B fragment (32x16 f16) for V_WMMA_F32_16X16X32_F16 from LDS-staged packed W.
// Packed layout: pw[r*N + n] = {W[2r][n], W[2r+1][n]} as half2 in one dword.
// Lane L, VGPR j needs K = kt*32 + (L>>4)*8 + 2j (+1) (+8 for j>=4) at column n = (L&15)+nt*16.
__device__ __forceinline__ v16h load_bfrag(const unsigned int* sW, int N, int kt,
                                           int nt, int lane) {
    const int n  = (lane & 15) + nt * 16;
    const int r0 = kt * 16 + ((lane >> 4) << 2);
    AFrag b;
#pragma unroll
    for (int j = 0; j < 8; ++j) b.u[j] = sW[(r0 + (j < 4 ? j : j + 4)) * N + n];
    return b.v;
}

// ---------------------------------------------------------------------------
// Prep: fold hyper-gate/bias into per-column (g, d): y = acc*g + d,
// with d = b*g + t*Wb. Layer 0 (d_fc1, din=1) folds further into (u, v):
// row(e)[k] = softplus(x[e]*u[k] + v[k]).
// ---------------------------------------------------------------------------
struct LayerP { const float *W, *b, *Wg, *bg, *Wb; int dout; };
struct PrepArgs { const float* t; LayerP L[8]; };

__global__ void k_prep(PrepArgs a, float* __restrict__ gates) {
    const float t = *a.t;
    const int n = threadIdx.x;            // 128 threads
    for (int L = 0; L < 8; ++L) {
        const LayerP p = a.L[L];
        if (n < p.dout) {
            const float g = 1.0f / (1.0f + __expf(-fmaf(t, p.Wg[n], p.bg[n])));
            const float c = t * p.Wb[n];
            float* o = gates + L * 256;
            if (L == 0) {
                o[n]       = p.W[n] * g;            // u
                o[HID + n] = fmaf(p.b[n], g, c);    // v
            } else {
                o[n]       = g;
                o[HID + n] = fmaf(p.b[n], g, c);    // d
            }
        }
    }
}

// Pack 6 weight matrices fp32 -> half2 dwords [K/2][N].
struct PackArgs { const float* W[6]; };

__global__ void k_pack(PackArgs a, unsigned int* __restrict__ pw) {
    const int K[6]   = {128, 128, 128, 128, 256, 128};
    const int N[6]   = {128, 128, 128, 128, 128, 64};
    const int OFF[6] = {0, 8192, 16384, 24576, 32768, 49152};
    int idx = blockIdx.x * 256 + threadIdx.x;
    for (int m = 0; m < 6; ++m) {
        const int cnt = (K[m] >> 1) * N[m];
        if (idx < cnt) {
            const int r = idx / N[m], n = idx % N[m];
            const float* W = a.W[m];
            pw[OFF[m] + idx] = pack2f16(W[(2 * r) * N[m] + n], W[(2 * r + 1) * N[m] + n]);
            return;
        }
        idx -= cnt;
    }
}

// ---------------------------------------------------------------------------
// GEMM 1: edge embedding. A generated on the fly from scalar x[e] (d_fc1
// fused), K=128, N=128; epilogue: (acc*g+d) * edge_attr -> f16.
// ---------------------------------------------------------------------------
__global__ __launch_bounds__(256) void k_gemm_dfc2(
    const float* __restrict__ X1, const float* __restrict__ EAT,
    const unsigned int* __restrict__ Wp,
    const float* __restrict__ uv, const float* __restrict__ gd,
    _Float16* __restrict__ EA16, int M)
{
    __shared__ unsigned int sW[64 * HID];   // 32KB
    __shared__ float sU[HID], sV[HID];
    stage_lds(sW, Wp, 64 * HID);
    if (threadIdx.x < HID) {
        sU[threadIdx.x] = uv[threadIdx.x];
        sV[threadIdx.x] = uv[HID + threadIdx.x];
    }
    __syncthreads();

    const int lane = threadIdx.x & 31, wave = threadIdx.x >> 5;
    const int row0 = blockIdx.x * 128 + wave * 16;
    const int m    = min(row0 + (lane & 15), M - 1);
    const float xv = X1[m];

    v8f acc[8] = {};
#pragma unroll
    for (int kt = 0; kt < 4; ++kt) {
        const int k0 = kt * 32 + ((lane >> 4) << 3);
        AFrag a;
#pragma unroll
        for (int i = 0; i < 16; ++i) {
            const int k = k0 + (i < 8 ? i : i + 8);
            a.h[i] = (_Float16)softplusf(fmaf(xv, sU[k], sV[k]));
        }
#pragma unroll
        for (int nt = 0; nt < 8; ++nt) {
            v16h b = load_bfrag(sW, HID, kt, nt, lane);
            acc[nt] = __builtin_amdgcn_wmma_f32_16x16x32_f16(
                false, a.v, false, b, (short)0, acc[nt], false, false);
        }
    }

    const int nlo = lane & 15;
    const int rb  = row0 + ((lane >> 4) << 3);
#pragma unroll
    for (int nt = 0; nt < 8; ++nt) {
        const int n = nlo + nt * 16;
        const float g = gd[n], d = gd[HID + n];
#pragma unroll
        for (int j = 0; j < 8; ++j) {
            const int mm = rb + j;
            if (mm < M) {
                float y = fmaf(acc[nt][j], g, d) * EAT[(size_t)mm * HID + n];
                EA16[(size_t)mm * HID + n] = (_Float16)y;
            }
        }
    }
}

// ---------------------------------------------------------------------------
// GEMM 2: node ConcatSquashLinear. X f16 [M,128] -> Y f16 [M,128].
// ---------------------------------------------------------------------------
__global__ __launch_bounds__(256) void k_gemm_node(
    const _Float16* __restrict__ X, const unsigned int* __restrict__ Wp,
    const float* __restrict__ gd, _Float16* __restrict__ Y, int M, int act)
{
    __shared__ unsigned int sW[64 * HID];   // 32KB
    stage_lds(sW, Wp, 64 * HID);
    __syncthreads();

    const int lane = threadIdx.x & 31, wave = threadIdx.x >> 5;
    const int row0 = blockIdx.x * 128 + wave * 16;
    const int m    = min(row0 + (lane & 15), M - 1);
    const _Float16* xr = X + (size_t)m * HID;
    __builtin_prefetch(X + (size_t)min(m + 128, M - 1) * HID, 0, 0);

    v8f acc[8] = {};
#pragma unroll
    for (int kt = 0; kt < 4; ++kt) {
        const int k0 = kt * 32 + ((lane >> 4) << 3);
        AFrag a;
        a.q[0] = *(const uint4*)(xr + k0);
        a.q[1] = *(const uint4*)(xr + k0 + 16);
#pragma unroll
        for (int nt = 0; nt < 8; ++nt) {
            v16h b = load_bfrag(sW, HID, kt, nt, lane);
            acc[nt] = __builtin_amdgcn_wmma_f32_16x16x32_f16(
                false, a.v, false, b, (short)0, acc[nt], false, false);
        }
    }

    const int nlo = lane & 15;
    const int rb  = row0 + ((lane >> 4) << 3);
#pragma unroll
    for (int nt = 0; nt < 8; ++nt) {
        const int n = nlo + nt * 16;
        const float g = gd[n], d = gd[HID + n];
#pragma unroll
        for (int j = 0; j < 8; ++j) {
            const int mm = rb + j;
            if (mm < M) {
                float y = fmaf(acc[nt][j], g, d);
                if (act) y = softplusf(y);
                Y[(size_t)mm * HID + n] = (_Float16)y;
            }
        }
    }
}

// ---------------------------------------------------------------------------
// GEMM 3: out_fc1, K=256. A built on the fly: k<128 -> h[src]*h[dst] (packed
// f16 mul), k>=128 -> edge_attr'. Never materializes pair_feat (saves ~820MB).
// ---------------------------------------------------------------------------
__global__ __launch_bounds__(256) void k_gemm_outfc1(
    const _Float16* __restrict__ H3, const _Float16* __restrict__ EA16,
    const int* __restrict__ src, const int* __restrict__ dst,
    const unsigned int* __restrict__ Wp, const float* __restrict__ gd,
    _Float16* __restrict__ PF1, int M)
{
    __shared__ unsigned int sW[128 * HID];  // 64KB (K=256)
    stage_lds(sW, Wp, 128 * HID);
    __syncthreads();

    const int lane = threadIdx.x & 31, wave = threadIdx.x >> 5;
    const int row0 = blockIdx.x * 128 + wave * 16;
    const int m    = min(row0 + (lane & 15), M - 1);
    const int se = src[m], de = dst[m];
    const _Float16* hs = H3 + (size_t)se * HID;
    const _Float16* hd = H3 + (size_t)de * HID;
    const _Float16* er = EA16 + (size_t)m * HID;

    v8f acc[8] = {};
#pragma unroll
    for (int kt = 0; kt < 8; ++kt) {
        const int k0 = kt * 32 + ((lane >> 4) << 3);
        AFrag a;
        if (kt < 4) {
            H8 p0, p1, q0, q1, r0, r1;
            p0.q = *(const uint4*)(hs + k0);
            p1.q = *(const uint4*)(hs + k0 + 16);
            q0.q = *(const uint4*)(hd + k0);
            q1.q = *(const uint4*)(hd + k0 + 16);
            r0.h = p0.h * q0.h;                 // v_pk_mul_f16
            r1.h = p1.h * q1.h;
            a.q[0] = r0.q; a.q[1] = r1.q;
        } else {
            const int kk = k0 - 128;
            a.q[0] = *(const uint4*)(er + kk);
            a.q[1] = *(const uint4*)(er + kk + 16);
        }
#pragma unroll
        for (int nt = 0; nt < 8; ++nt) {
            v16h b = load_bfrag(sW, HID, kt, nt, lane);
            acc[nt] = __builtin_amdgcn_wmma_f32_16x16x32_f16(
                false, a.v, false, b, (short)0, acc[nt], false, false);
        }
    }

    const int nlo = lane & 15;
    const int rb  = row0 + ((lane >> 4) << 3);
#pragma unroll
    for (int nt = 0; nt < 8; ++nt) {
        const int n = nlo + nt * 16;
        const float g = gd[n], d = gd[HID + n];
#pragma unroll
        for (int j = 0; j < 8; ++j) {
            const int mm = rb + j;
            if (mm < M)
                PF1[(size_t)mm * HID + n] = (_Float16)softplusf(fmaf(acc[nt][j], g, d));
        }
    }
}

// ---------------------------------------------------------------------------
// GEMM 4: out_fc2, K=128, N=64.
// ---------------------------------------------------------------------------
__global__ __launch_bounds__(256) void k_gemm_outfc2(
    const _Float16* __restrict__ X, const unsigned int* __restrict__ Wp,
    const float* __restrict__ gd, _Float16* __restrict__ Y, int M)
{
    __shared__ unsigned int sW[64 * 64];    // 16KB
    stage_lds(sW, Wp, 64 * 64);
    __syncthreads();

    const int lane = threadIdx.x & 31, wave = threadIdx.x >> 5;
    const int row0 = blockIdx.x * 128 + wave * 16;
    const int m    = min(row0 + (lane & 15), M - 1);
    const _Float16* xr = X + (size_t)m * HID;

    v8f acc[4] = {};
#pragma unroll
    for (int kt = 0; kt < 4; ++kt) {
        const int k0 = kt * 32 + ((lane >> 4) << 3);
        AFrag a;
        a.q[0] = *(const uint4*)(xr + k0);
        a.q[1] = *(const uint4*)(xr + k0 + 16);
#pragma unroll
        for (int nt = 0; nt < 4; ++nt) {
            v16h b = load_bfrag(sW, 64, kt, nt, lane);
            acc[nt] = __builtin_amdgcn_wmma_f32_16x16x32_f16(
                false, a.v, false, b, (short)0, acc[nt], false, false);
        }
    }

    const int nlo = lane & 15;
    const int rb  = row0 + ((lane >> 4) << 3);
#pragma unroll
    for (int nt = 0; nt < 4; ++nt) {
        const int n = nlo + nt * 16;
        const float g = gd[n], d = gd[HID + n];
#pragma unroll
        for (int j = 0; j < 8; ++j) {
            const int mm = rb + j;
            if (mm < M)
                Y[(size_t)mm * 64 + n] = (_Float16)softplusf(fmaf(acc[nt][j], g, d));
        }
    }
}

// ---------------------------------------------------------------------------
// GINE message + scatter-add: agg[dst] += softplus(h[src] + ea). 4 cols/thread.
// ---------------------------------------------------------------------------
__global__ __launch_bounds__(256) void k_msg_scatter(
    const _Float16* __restrict__ H, const _Float16* __restrict__ EA,
    const int* __restrict__ src, const int* __restrict__ dst,
    float* __restrict__ agg, int E)
{
    const long t = (long)blockIdx.x * 256 + threadIdx.x;
    if (t >= (long)E * 32) return;
    const int e = (int)(t >> 5);
    const int c = (int)(t & 31) << 2;
    const int s = src[e], d = dst[e];
    H4 a, b;
    a.q = *(const uint2*)(H  + (size_t)s * HID + c);
    b.q = *(const uint2*)(EA + (size_t)e * HID + c);
    float* ap = agg + (size_t)d * HID + c;
#pragma unroll
    for (int i = 0; i < 4; ++i) {
        const float v = softplusf((float)a.h[i] + (float)b.h[i]);
        __hip_atomic_fetch_add(ap + i, v, __ATOMIC_RELAXED, __HIP_MEMORY_SCOPE_AGENT);
    }
}

__global__ void k_f32_to_f16(const float* __restrict__ in, _Float16* __restrict__ out, long n) {
    const long i = (long)blockIdx.x * 256 + threadIdx.x;
    if (i < n) out[i] = (_Float16)in[i];
}

__global__ void k_add_cvt(const float* __restrict__ agg, const _Float16* __restrict__ h,
                          _Float16* __restrict__ out, long n) {
    const long i = (long)blockIdx.x * 256 + threadIdx.x;
    if (i < n) out[i] = (_Float16)(agg[i] + (float)h[i]);
}

// Final: out[e] = (pf2[e,:] . W3) * g + d
__global__ __launch_bounds__(256) void k_final(
    const _Float16* __restrict__ PF2, const float* __restrict__ W3,
    const float* __restrict__ gd, float* __restrict__ out, int E)
{
    __shared__ float sw[64];
    __shared__ float sgd[2];
    if (threadIdx.x < 64) sw[threadIdx.x] = W3[threadIdx.x];
    if (threadIdx.x < 2)  sgd[threadIdx.x] = gd[threadIdx.x * HID];
    __syncthreads();
    const int e = blockIdx.x * 256 + threadIdx.x;
    if (e >= E) return;
    const _Float16* r = PF2 + (size_t)e * 64;
    float acc = 0.0f;
#pragma unroll
    for (int i = 0; i < 64; ++i) acc = fmaf((float)r[i], sw[i], acc);
    out[e] = fmaf(acc, sgd[0], sgd[1]);
}

// ---------------------------------------------------------------------------
extern "C" void kernel_launch(void* const* d_in, const int* in_sizes, int n_in,
                              void* d_out, int out_size, void* d_ws, size_t ws_size,
                              hipStream_t stream) {
    (void)n_in; (void)out_size; (void)ws_size;
    const float* t_p  = (const float*)d_in[0];
    const float* x_p  = (const float*)d_in[1];
    const float* nat  = (const float*)d_in[2];
    const float* eat  = (const float*)d_in[3];
    const int*   eidx = (const int*)d_in[4];
    const int E = in_sizes[1];
    const int N = in_sizes[2] / HID;
    const int* src = eidx;
    const int* dst = eidx + E;

    // params (insertion order): d_fc1, d_fc2, conv1..3, out_fc1..3; each W,b,Wg,bg,Wb
    auto Lp = [&](int base, int dout) {
        LayerP p;
        p.W  = (const float*)d_in[base + 0];
        p.b  = (const float*)d_in[base + 1];
        p.Wg = (const float*)d_in[base + 2];
        p.bg = (const float*)d_in[base + 3];
        p.Wb = (const float*)d_in[base + 4];
        p.dout = dout;
        return p;
    };
    PrepArgs pa;
    pa.t = t_p;
    pa.L[0] = Lp(5, 128);   // d_fc1
    pa.L[1] = Lp(10, 128);  // d_fc2
    pa.L[2] = Lp(15, 128);  // conv1
    pa.L[3] = Lp(20, 128);  // conv2
    pa.L[4] = Lp(25, 128);  // conv3
    pa.L[5] = Lp(30, 128);  // out_fc1
    pa.L[6] = Lp(35, 64);   // out_fc2
    pa.L[7] = Lp(40, 1);    // out_fc3

    PackArgs pk;
    pk.W[0] = (const float*)d_in[10]; // d_fc2 W [128,128]
    pk.W[1] = (const float*)d_in[15]; // conv1
    pk.W[2] = (const float*)d_in[20]; // conv2
    pk.W[3] = (const float*)d_in[25]; // conv3
    pk.W[4] = (const float*)d_in[30]; // out_fc1 [256,128]
    pk.W[5] = (const float*)d_in[35]; // out_fc2 [128,64]

    // workspace layout
    const size_t MB = 1ull << 20;
    char* ws = (char*)d_ws;
    float*        gates = (float*)(ws);                     //  8KB
    unsigned int* pw    = (unsigned int*)(ws + 64 * 1024);  // 208KB packed W
    _Float16* EA16 = (_Float16*)(ws + 1 * MB);              // E*128 f16 (205MB)
    _Float16* PF1  = (_Float16*)(ws + (1 + 208) * MB);      // E*128 f16
    _Float16* PF2  = (_Float16*)(ws + (1 + 416) * MB);      // E*64  f16 (103MB)
    _Float16* H16A = (_Float16*)(ws + (1 + 520) * MB);      // N*128 f16
    _Float16* H16B = (_Float16*)(ws + (1 + 533) * MB);
    _Float16* XIN  = (_Float16*)(ws + (1 + 546) * MB);
    float*    AGG  = (float*)   (ws + (1 + 559) * MB);      // N*128 f32 (26MB)

    const unsigned int* PW_DFC2  = pw;
    const unsigned int* PW_CONV1 = pw + 8192;
    const unsigned int* PW_CONV2 = pw + 16384;
    const unsigned int* PW_CONV3 = pw + 24576;
    const unsigned int* PW_OUT1  = pw + 32768;
    const unsigned int* PW_OUT2  = pw + 49152;

    const long nodeElems = (long)N * HID;
    const int edgeBlocks = (E + 127) / 128;                 // 6250
    const int nodeBlocks = (N + 127) / 128;                 // 391
    const int cvtBlocks  = (int)((nodeElems + 255) / 256);
    const int scatBlocks = (int)(((long)E * 32 + 255) / 256);

    k_prep<<<1, 128, 0, stream>>>(pa, gates);
    k_pack<<<(53248 + 255) / 256, 256, 0, stream>>>(pk, pw);
    k_f32_to_f16<<<cvtBlocks, 256, 0, stream>>>(nat, H16A, nodeElems);

    // edge embedding (d_fc1 fused) * edge_attr -> EA16
    k_gemm_dfc2<<<edgeBlocks, 256, 0, stream>>>(x_p, eat, PW_DFC2,
                                                gates + 0 * 256, gates + 1 * 256, EA16, E);

    // conv1
    (void)hipMemsetAsync(AGG, 0, (size_t)nodeElems * sizeof(float), stream);
    k_msg_scatter<<<scatBlocks, 256, 0, stream>>>(H16A, EA16, src, dst, AGG, E);
    k_add_cvt<<<cvtBlocks, 256, 0, stream>>>(AGG, H16A, XIN, nodeElems);
    k_gemm_node<<<nodeBlocks, 256, 0, stream>>>(XIN, PW_CONV1, gates + 2 * 256, H16B, N, 1);

    // conv2
    (void)hipMemsetAsync(AGG, 0, (size_t)nodeElems * sizeof(float), stream);
    k_msg_scatter<<<scatBlocks, 256, 0, stream>>>(H16B, EA16, src, dst, AGG, E);
    k_add_cvt<<<cvtBlocks, 256, 0, stream>>>(AGG, H16B, XIN, nodeElems);
    k_gemm_node<<<nodeBlocks, 256, 0, stream>>>(XIN, PW_CONV2, gates + 3 * 256, H16A, N, 1);

    // conv3 (no activation)
    (void)hipMemsetAsync(AGG, 0, (size_t)nodeElems * sizeof(float), stream);
    k_msg_scatter<<<scatBlocks, 256, 0, stream>>>(H16A, EA16, src, dst, AGG, E);
    k_add_cvt<<<cvtBlocks, 256, 0, stream>>>(AGG, H16A, XIN, nodeElems);
    k_gemm_node<<<nodeBlocks, 256, 0, stream>>>(XIN, PW_CONV3, gates + 4 * 256, H16B, N, 0);

    // pair MLP: out_fc1 (pair_feat fused into A-fragments), out_fc2, out_fc3
    k_gemm_outfc1<<<edgeBlocks, 256, 0, stream>>>(H16B, EA16, src, dst, PW_OUT1,
                                                  gates + 5 * 256, PF1, E);
    k_gemm_outfc2<<<edgeBlocks, 256, 0, stream>>>(PF1, PW_OUT2, gates + 6 * 256, PF2, E);
    k_final<<<(E + 255) / 256, 256, 0, stream>>>(PF2, (const float*)d_in[40],
                                                 gates + 7 * 256, (float*)d_out, E);
}
